// CompLayer_50448685859245
// MI455X (gfx1250) — matrix operand order repaired and checked by reading, not compiled, not run on hardware
//
#include <hip/hip_runtime.h>
#include <hip/hip_fp16.h>

typedef __attribute__((ext_vector_type(16))) _Float16 v16h;
typedef __attribute__((ext_vector_type(8)))  _Float16 v8h;
typedef __attribute__((ext_vector_type(8)))  float    v8f;

#define N_NODES 30000
#define DEG     32
#define HDIM    256
#define KTOP    15

// ---------------------------------------------------------------------------
// Kernel 1: W (HxH, row-major, f32) -> Wt (HxH, column-of-W per row, f16)
// Wt[n][k] = W[k][n]; gives each WMMA B-lane a contiguous K run.
// ---------------------------------------------------------------------------
__global__ __launch_bounds__(256) void
wconv_kernel(const float* __restrict__ W, _Float16* __restrict__ WtH)
{
    const int n = blockIdx.x;       // output column of W == row of Wt
    const int k = threadIdx.x;
    WtH[n * HDIM + k] = (_Float16)W[k * HDIM + n];
}

// ---------------------------------------------------------------------------
// Kernel 2: per-node edge phase. One block (8 waves, 256 thr) per node.
//  - gather comp = ent[src] + rel[rid], score via wave32 shuffle reduction
//  - softmax over D=32, exact top-15 mask (rank = #greater + #equal&lower idx)
//  - weighted sum -> neigh, stored as f16 for the WMMA GEMM
// ---------------------------------------------------------------------------
__global__ __launch_bounds__(256) void
edge_phase_kernel(const float* __restrict__ ent, const float* __restrict__ rel,
                  const int* __restrict__ src,  const int* __restrict__ rid,
                  _Float16* __restrict__ neighH)
{
    __shared__ float s_comp[DEG][HDIM];   // 32 KB
    __shared__ float s_hdst[HDIM];
    __shared__ float s_score[DEG];
    __shared__ float s_w[DEG];

    const int n    = blockIdx.x;
    const int t    = threadIdx.x;
    const int wave = t >> 5;              // wave32
    const int lane = t & 31;
    const int h0   = lane * 8;            // each lane owns 8 h-values

    s_hdst[t] = ent[n * HDIM + t];
    __syncthreads();

    float hd[8];
    #pragma unroll
    for (int k = 0; k < 8; ++k) hd[k] = s_hdst[h0 + k];

    // each wave handles 4 of the 32 incoming edges
    #pragma unroll
    for (int j = 0; j < 4; ++j) {
        const int d = wave * 4 + j;
        const int s = src[n * DEG + d];
        const int r = rid[n * DEG + d];
        const float4* ep = (const float4*)(ent + s * HDIM + h0);
        const float4* rp = (const float4*)(rel + r * HDIM + h0);
        float4 e0 = ep[0], e1 = ep[1];
        float4 r0 = rp[0], r1 = rp[1];
        float c[8] = { e0.x + r0.x, e0.y + r0.y, e0.z + r0.z, e0.w + r0.w,
                       e1.x + r1.x, e1.y + r1.y, e1.z + r1.z, e1.w + r1.w };
        float p = 0.f;
        #pragma unroll
        for (int k = 0; k < 8; ++k) p = fmaf(c[k], hd[k], p);
        #pragma unroll
        for (int off = 16; off > 0; off >>= 1) p += __shfl_xor(p, off, 32);
        if (lane == 0) s_score[d] = p;
        float4* cp = (float4*)&s_comp[d][h0];
        cp[0] = make_float4(c[0], c[1], c[2], c[3]);
        cp[1] = make_float4(c[4], c[5], c[6], c[7]);
    }
    __syncthreads();

    // wave 0: softmax over D=32 + exact top-15 selection
    if (wave == 0) {
        float sc = s_score[lane];
        float m = sc;
        #pragma unroll
        for (int off = 16; off > 0; off >>= 1) m = fmaxf(m, __shfl_xor(m, off, 32));
        float e = __expf(sc - m);
        float sum = e;
        #pragma unroll
        for (int off = 16; off > 0; off >>= 1) sum += __shfl_xor(sum, off, 32);
        const float nrm = e / sum;
        int rank = 0;
        #pragma unroll
        for (int j2 = 0; j2 < 32; ++j2) {
            float vj = __shfl(nrm, j2, 32);
            rank += (vj > nrm) || (vj == nrm && j2 < lane);
        }
        s_w[lane] = (rank < KTOP) ? nrm : 0.f;
    }
    __syncthreads();

    float acc = 0.f;
    #pragma unroll
    for (int d = 0; d < DEG; ++d) acc = fmaf(s_w[d], s_comp[d][t], acc);
    neighH[n * HDIM + t] = (_Float16)acc;
}

// ---------------------------------------------------------------------------
// Kernel 3: out = tanh(neigh @ W) via v_wmma_f32_16x16x32_f16.
// One 16x16 tile per wave; 8 waves/block; K loop = 256/32 = 8 WMMAs.
// Fragment layouts used here (contiguous k order, same in A and B):
//   A: lane m = lane&15; halves: K in [kk+16*kh, +16)
//   B: lane n = lane&15; halves: K in [kk+16*kh, +16) from Wt row n
//   D: VGPR r -> (M = r + 8*kh, N = lane&15)
// ---------------------------------------------------------------------------
__global__ __launch_bounds__(256) void
gemm_tanh_kernel(const _Float16* __restrict__ A, const _Float16* __restrict__ Bt,
                 float* __restrict__ out)
{
    const int t     = threadIdx.x;
    const int wave  = t >> 5;
    const int lane  = t & 31;
    const int mtile = blockIdx.x >> 1;                  // 1875 M-tiles
    const int ntile = ((blockIdx.x & 1) << 3) + wave;   // 16 N-tiles
    const int kh    = lane >> 4;

    const _Float16* arow = A  + (mtile * 16 + (lane & 15)) * HDIM;
    const _Float16* brow = Bt + (ntile * 16 + (lane & 15)) * HDIM;

    v8f c = {};
    #pragma unroll
    for (int it = 0; it < 8; ++it) {
        const int kk = it * 32;
        v16h a = *(const v16h*)(arow + kk + 16 * kh);
        v16h b = *(const v16h*)(brow + kk + 16 * kh);
        c = __builtin_amdgcn_wmma_f32_16x16x32_f16(
                /*neg_a=*/false, a, /*neg_b=*/false, b,
                /*c_mod=*/(short)0, c, /*reuse_a=*/false, /*reuse_b=*/false);
    }

    const int ncol = ntile * 16 + (lane & 15);
    #pragma unroll
    for (int r = 0; r < 8; ++r) {
        const int m = mtile * 16 + 8 * kh + r;
        out[m * HDIM + ncol] = tanhf(c[r]);
    }
}

// ---------------------------------------------------------------------------
extern "C" void kernel_launch(void* const* d_in, const int* in_sizes, int n_in,
                              void* d_out, int out_size, void* d_ws, size_t ws_size,
                              hipStream_t stream)
{
    const float* ent = (const float*)d_in[0];   // [30000, 256]
    const float* rel = (const float*)d_in[1];   // [474, 256]
    const float* W   = (const float*)d_in[2];   // [256, 256]
    const int*   src = (const int*)  d_in[3];   // [30000, 32]
    const int*   rid = (const int*)  d_in[4];   // [30000, 32]
    float*       out = (float*)d_out;           // [30000, 256]

    _Float16* WtH    = (_Float16*)d_ws;                                   // 128 KB
    _Float16* neighH = (_Float16*)((char*)d_ws + HDIM * HDIM * sizeof(_Float16));

    wconv_kernel<<<HDIM, HDIM, 0, stream>>>(W, WtH);
    edge_phase_kernel<<<N_NODES, 256, 0, stream>>>(ent, rel, src, rid, neighH);
    gemm_tanh_kernel<<<N_NODES / 8, 256, 0, stream>>>(neighH, WtH, out);
}
